// GraphSAGE_46686294507955
// MI455X (gfx1250) — compile-verified, matrix operand order checked by reading, not run on hardware
//
#include <hip/hip_runtime.h>
#include <hip/hip_bf16.h>

// ---------------------------------------------------------------------------
// GraphSAGE (LSTM aggregator) for MI455X / gfx1250.
//   N=50000 nodes, D=32 neighbors, FIN=5, HID=32.
// Layer-2 LSTM dominates (13.1 G MACs) and is mapped onto
// v_wmma_f32_16x16x32_bf16: one wave = 16 nodes, 16 WMMAs per timestep.
// Gate nonlinearities use hardware v_tanh_f32; the neighbor gather is
// software-pipelined one timestep ahead of the WMMA consumption.
// ---------------------------------------------------------------------------

#define NN   50000
#define DD   32
#define FINN 5
#define HIDD 32
#define NT   (NN / 16)   // 3125 node tiles

typedef __attribute__((ext_vector_type(16))) __bf16 v16bf;
typedef __attribute__((ext_vector_type(8)))  __bf16 v8bf;
typedef __attribute__((ext_vector_type(8)))  float  v8f;

union BF16x16 { v16bf v; v8bf h[2]; };

__device__ inline v8f wmma_bf16(v16bf a, v16bf b, v8f c) {
    // 8 args: (neg_a, A, neg_b, B, c_mod, C, reuse_a, reuse_b)
    return __builtin_amdgcn_wmma_f32_16x16x32_bf16(
        false, a, false, b, (short)0, c, false, false);
}

// Branchless transcendentals: hardware v_tanh_f32 (gfx1250 TRANS op),
// fallback to native v_exp_f32 + v_rcp_f32 (branch-free, saturates cleanly).
__device__ inline float fast_tanh(float x) {
#if __has_builtin(__builtin_amdgcn_tanhf)
    return __builtin_amdgcn_tanhf(x);
#else
    return 1.0f - 2.0f * __builtin_amdgcn_rcpf(__expf(2.0f * x) + 1.0f);
#endif
}

__device__ inline float sigm(float x) {
#if __has_builtin(__builtin_amdgcn_tanhf)
    return 0.5f * __builtin_amdgcn_tanhf(0.5f * x) + 0.5f;
#else
    return __builtin_amdgcn_rcpf(1.0f + __expf(-x));
#endif
}

// Load one 16-lane fragment (A or B side) from a row-major bf16 matrix with
// row stride 32 elements.  16-bit 16x32 fragment layout:
//   lanes 0-15 : row = lane,    K = {0..7, 16..23}
//   lanes 16-31: row = lane-16, K = {8..15, 24..31}
__device__ inline v16bf load_frag(const __bf16* base, int row, int hi) {
    BF16x16 r;
    const __bf16* p = base + row * 32 + hi * 8;
    r.h[0] = *reinterpret_cast<const v8bf*>(p);
    r.h[1] = *reinterpret_cast<const v8bf*>(p + 16);
    return r.v;
}

__device__ inline v16bf zero16() {
    BF16x16 z;
#pragma unroll
    for (int i = 0; i < 16; ++i) z.v[i] = (__bf16)0.0f;
    return z.v;
}

__device__ inline v8f splat8(float b) {
    v8f r = {b, b, b, b, b, b, b, b};
    return r;
}

// ---------------------------------------------------------------------------
// Prep: convert layer-2 / head weights (f32 -> bf16) into workspace.
// Layout in wts (bf16 elements): Wih[4096] Whh[4096] Wself[1024] Wneigh[1024] fc1[512]
// ---------------------------------------------------------------------------
__global__ void prep_kernel(const float* __restrict__ Wih,
                            const float* __restrict__ Whh,
                            const float* __restrict__ Wself,
                            const float* __restrict__ Wneigh,
                            const float* __restrict__ fc1,
                            __bf16* __restrict__ wts) {
    int i = blockIdx.x * blockDim.x + threadIdx.x;
    if (i < 4096)        wts[i] = (__bf16)Wih[i];
    else if (i < 8192)   wts[i] = (__bf16)Whh[i - 4096];
    else if (i < 9216)   wts[i] = (__bf16)Wself[i - 8192];
    else if (i < 10240)  wts[i] = (__bf16)Wneigh[i - 9216];
    else if (i < 10752)  wts[i] = (__bf16)fc1[i - 10240];
}

// ---------------------------------------------------------------------------
// Layer 1: hid=5 LSTM + projection to 32, one thread per node (scalar VALU).
// Emits h1 in bf16, ready for layer-2 WMMA fragments.
// ---------------------------------------------------------------------------
__global__ void l1_kernel(const float* __restrict__ x,
                          const int* __restrict__ nidx,
                          const float* __restrict__ Wih, const float* __restrict__ Whh,
                          const float* __restrict__ bih, const float* __restrict__ bhh,
                          const float* __restrict__ Wself, const float* __restrict__ Wneigh,
                          const float* __restrict__ bo,
                          __bf16* __restrict__ h1) {
    __shared__ float sWih[100], sWhh[100], sB[20], sWs[160], sWn[160], sBo[32];
    int tid = threadIdx.x;
    for (int i = tid; i < 100; i += blockDim.x) { sWih[i] = Wih[i]; sWhh[i] = Whh[i]; }
    for (int i = tid; i < 20;  i += blockDim.x) sB[i] = bih[i] + bhh[i];
    for (int i = tid; i < 160; i += blockDim.x) { sWs[i] = Wself[i]; sWn[i] = Wneigh[i]; }
    for (int i = tid; i < 32;  i += blockDim.x) sBo[i] = bo[i];
    __syncthreads();

    int node = blockIdx.x * blockDim.x + tid;
    if (node >= NN) return;

    float xr[FINN];
#pragma unroll
    for (int k = 0; k < FINN; ++k) xr[k] = x[node * FINN + k];

    float h[FINN] = {0, 0, 0, 0, 0}, c[FINN] = {0, 0, 0, 0, 0};
    for (int t = 0; t < DD; ++t) {
        int id = nidx[node * DD + t];
        float xt[FINN];
#pragma unroll
        for (int k = 0; k < FINN; ++k) xt[k] = x[id * FINN + k];
        float g[20];
#pragma unroll
        for (int r = 0; r < 20; ++r) {
            float s = sB[r];
#pragma unroll
            for (int k = 0; k < FINN; ++k)
                s += sWih[r * FINN + k] * xt[k] + sWhh[r * FINN + k] * h[k];
            g[r] = s;
        }
#pragma unroll
        for (int k = 0; k < FINN; ++k) {
            float cn = sigm(g[5 + k]) * c[k] + sigm(g[k]) * fast_tanh(g[10 + k]);
            c[k] = cn;
            h[k] = sigm(g[15 + k]) * fast_tanh(cn);
        }
    }
#pragma unroll
    for (int n = 0; n < HIDD; ++n) {
        float s = sBo[n];
#pragma unroll
        for (int k = 0; k < FINN; ++k)
            s += sWs[n * FINN + k] * xr[k] + sWn[n * FINN + k] * h[k];
        s = s > 0.0f ? s : 0.0f;                 // ReLU
        h1[node * HIDD + n] = (__bf16)s;
    }
}

// ---------------------------------------------------------------------------
// Layer 2 + head: one wave per 16-node tile, WMMA LSTM recurrence.
// Block = 128 threads = 4 independent waves.
// ---------------------------------------------------------------------------
__global__ void __launch_bounds__(128)
l2_kernel(const __bf16* __restrict__ h1, const int* __restrict__ nidx,
          const __bf16* __restrict__ wts,
          const float* __restrict__ bih, const float* __restrict__ bhh,
          const float* __restrict__ l2b, const float* __restrict__ fc1b,
          const float* __restrict__ fc2W, const float* __restrict__ fc2b,
          float* __restrict__ out) {
    __shared__ int s_idx[4][16 * 33];                      // padded: conflict-free
    __shared__ __align__(16) __bf16 s_h[4][16 * 32];       // C/D -> A relayout buffer

    const int wid  = threadIdx.x >> 5;
    const int lane = threadIdx.x & 31;
    const int tile = blockIdx.x * 4 + wid;
    if (tile >= NT) return;                                // wave-uniform exit
    const int base = tile * 16;
    const int r16  = lane & 15;
    const int hi   = lane >> 4;

    // Stage the 16x32 neighbor-index block for this tile into LDS.
    for (int r = 0; r < 16; ++r)
        s_idx[wid][r * 33 + lane] = nidx[(base + r) * DD + lane];

    const __bf16* Wih = wts;
    const __bf16* Whh = wts + 4096;
    const __bf16* Wsf = wts + 8192;
    const __bf16* Wng = wts + 9216;
    const __bf16* Wf1 = wts + 10240;

    // Keep all 16 weight B-fragments resident in VGPRs across the recurrence.
    v16bf Bih[8], Bhh[8];
    float bias[8];
#pragma unroll
    for (int j = 0; j < 8; ++j) {
        Bih[j]  = load_frag(Wih, j * 16 + r16, hi);
        Bhh[j]  = load_frag(Whh, j * 16 + r16, hi);
        bias[j] = bih[j * 16 + r16] + bhh[j * 16 + r16];
    }

    v16bf hA = zero16();
    v8f c0 = {0, 0, 0, 0, 0, 0, 0, 0};
    v8f c1 = {0, 0, 0, 0, 0, 0, 0, 0};
    asm volatile("s_wait_dscnt 0" ::: "memory");

    // Software-pipelined gather: fetch timestep t+1 while computing t.
    int nid0 = s_idx[wid][r16 * 33 + 0];
    v16bf xA = load_frag(h1, nid0, hi);

    for (int t = 0; t < DD; ++t) {
        v16bf xCur = xA;
        if (t + 1 < DD) {
            int nn = s_idx[wid][r16 * 33 + (t + 1)];
            xA = load_frag(h1, nn, hi);                    // in flight during WMMAs
        }
#pragma unroll
        for (int p = 0; p < 2; ++p) {                      // hidden halves 0-15 / 16-31
            v8f ai = splat8(bias[0 + p]);
            v8f af = splat8(bias[2 + p]);
            v8f ag = splat8(bias[4 + p]);
            v8f ao = splat8(bias[6 + p]);
            ai = wmma_bf16(xCur, Bih[0 + p], ai); ai = wmma_bf16(hA, Bhh[0 + p], ai);
            af = wmma_bf16(xCur, Bih[2 + p], af); af = wmma_bf16(hA, Bhh[2 + p], af);
            ag = wmma_bf16(xCur, Bih[4 + p], ag); ag = wmma_bf16(hA, Bhh[4 + p], ag);
            ao = wmma_bf16(xCur, Bih[6 + p], ao); ao = wmma_bf16(hA, Bhh[6 + p], ao);
            v8f& cc = p ? c1 : c0;
#pragma unroll
            for (int e = 0; e < 8; ++e) {                  // cell update on C/D layout
                float cn = sigm(af[e]) * cc[e] + sigm(ai[e]) * fast_tanh(ag[e]);
                cc[e] = cn;
                float hv = sigm(ao[e]) * fast_tanh(cn);
                s_h[wid][(e + 8 * hi) * 32 + p * 16 + r16] = (__bf16)hv;
            }
        }
        asm volatile("s_wait_dscnt 0" ::: "memory");       // DS in-order per wave
        hA = load_frag(&s_h[wid][0], r16, hi);             // C/D -> A relayout
    }

    // out = relu( h1_self @ Wself^T + h_n @ Wneigh^T + b )
    v16bf selfA = load_frag(h1, base + r16, hi);
    v16bf Bs0 = load_frag(Wsf, r16, hi);
    v16bf Bs1 = load_frag(Wsf, 16 + r16, hi);
    v16bf Bn0 = load_frag(Wng, r16, hi);
    v16bf Bn1 = load_frag(Wng, 16 + r16, hi);
    v8f o0 = splat8(l2b[r16]);
    v8f o1 = splat8(l2b[16 + r16]);
    o0 = wmma_bf16(selfA, Bs0, o0); o0 = wmma_bf16(hA, Bn0, o0);
    o1 = wmma_bf16(selfA, Bs1, o1); o1 = wmma_bf16(hA, Bn1, o1);
#pragma unroll
    for (int e = 0; e < 8; ++e) {
        float a = o0[e]; a = a > 0.0f ? a : 0.0f;
        float b = o1[e]; b = b > 0.0f ? b : 0.0f;
        int m = e + 8 * hi;
        s_h[wid][m * 32 + r16]      = (__bf16)a;
        s_h[wid][m * 32 + 16 + r16] = (__bf16)b;
    }
    asm volatile("s_wait_dscnt 0" ::: "memory");

    // head: fc1 (16x32 @ 32x16) via WMMA, leaky-relu, fc2 via shfl butterfly
    v16bf oA = load_frag(&s_h[wid][0], r16, hi);
    v16bf Bf = load_frag(Wf1, r16, hi);
    v8f f1 = splat8(fc1b[r16]);
    f1 = wmma_bf16(oA, Bf, f1);

    float w2 = fc2W[r16];
    float b2 = fc2b[0];
#pragma unroll
    for (int e = 0; e < 8; ++e) {
        float v = f1[e];
        v = v > 0.0f ? v : 0.01f * v;                      // LeakyReLU(0.01)
        v *= w2;
        v += __shfl_xor(v, 1, 32);                         // reduce 16 lanes (xor
        v += __shfl_xor(v, 2, 32);                         //  masks stay inside a
        v += __shfl_xor(v, 4, 32);                         //  16-lane half)
        v += __shfl_xor(v, 8, 32);
        if (r16 == 0) out[base + e + 8 * hi] = v + b2;
    }
}

// ---------------------------------------------------------------------------
extern "C" void kernel_launch(void* const* d_in, const int* in_sizes, int n_in,
                              void* d_out, int out_size, void* d_ws, size_t ws_size,
                              hipStream_t stream) {
    const float* x        = (const float*)d_in[0];
    const int*   nidx     = (const int*)d_in[1];
    const float* l1_Wih   = (const float*)d_in[2];
    const float* l1_Whh   = (const float*)d_in[3];
    const float* l1_bih   = (const float*)d_in[4];
    const float* l1_bhh   = (const float*)d_in[5];
    const float* l1_Wself = (const float*)d_in[6];
    const float* l1_Wneigh= (const float*)d_in[7];
    const float* l1_b     = (const float*)d_in[8];
    const float* l2_Wih   = (const float*)d_in[9];
    const float* l2_Whh   = (const float*)d_in[10];
    const float* l2_bih   = (const float*)d_in[11];
    const float* l2_bhh   = (const float*)d_in[12];
    const float* l2_Wself = (const float*)d_in[13];
    const float* l2_Wneigh= (const float*)d_in[14];
    const float* l2_b     = (const float*)d_in[15];
    const float* fc1_W    = (const float*)d_in[16];
    const float* fc1_b    = (const float*)d_in[17];
    const float* fc2_W    = (const float*)d_in[18];
    const float* fc2_b    = (const float*)d_in[19];

    __bf16* h1  = (__bf16*)d_ws;                 // N*32 bf16 = 3.2 MB
    __bf16* wts = h1 + (size_t)NN * HIDD;        // 10752 bf16 weights

    prep_kernel<<<(10752 + 255) / 256, 256, 0, stream>>>(
        l2_Wih, l2_Whh, l2_Wself, l2_Wneigh, fc1_W, wts);
    l1_kernel<<<(NN + 255) / 256, 256, 0, stream>>>(
        x, nidx, l1_Wih, l1_Whh, l1_bih, l1_bhh, l1_Wself, l1_Wneigh, l1_b, h1);
    l2_kernel<<<(NT + 3) / 4, 128, 0, stream>>>(
        h1, nidx, wts, l2_bih, l2_bhh, l2_b, fc1_b, fc2_W, fc2_b, (float*)d_out);
}
